// IS_MultiHeadAttention_64647847739754
// MI455X (gfx1250) — matrix operand order repaired and checked by reading, not compiled, not run on hardware
//
#include <hip/hip_runtime.h>

#define DEV static __device__ __forceinline__

typedef __bf16 bf16;
typedef __attribute__((ext_vector_type(16))) __bf16 v16bf;
typedef __attribute__((ext_vector_type(8)))  __bf16 v8bf;
typedef __attribute__((ext_vector_type(8)))  float  v8f;

constexpr int B_   = 8;
constexpr int S_   = 384;
constexpr int HID_ = 512;
constexpr int NH_  = 8;
constexpr int HD_  = 64;              // HID/NH
constexpr int BS_  = B_ * S_;         // 3072 (total rows / total keys)
constexpr int SS_  = S_ * S_;         // 147456
constexpr float INV_SCALE = 0.04419417382415922f;  // 1/sqrt(512)
constexpr float EPS_ = 1e-6f;

DEV v8f wmma_bf16(v16bf a, v16bf b, v8f c) {
  return __builtin_amdgcn_wmma_f32_16x16x32_bf16(false, a, false, b, (short)0, c,
                                                 false, false);
}

// ---- fragment loaders ---------------------------------------------------
// A fragment (16x32 bf16): lane holds row m=l&15; K(e) = (e%8)+16*(e/8)+8*hi.
// rp points at row start + k0.
DEV v16bf load_fragA_bf16(const bf16* rp, int hi) {
  const bf16* p = rp + 8 * hi;
  v8bf lo = *(const v8bf*)(p);
  v8bf hh = *(const v8bf*)(p + 16);
  v16bf f;
#pragma unroll
  for (int i = 0; i < 8; ++i) { f[i] = lo[i]; f[i + 8] = hh[i]; }
  return f;
}

// A fragment sourced from f32 memory (converted to bf16 on load).
DEV v16bf load_fragA_f32(const float* rp, int hi) {
  const float* p = rp + 8 * hi;
  v16bf f;
#pragma unroll
  for (int i = 0; i < 8; ++i) { f[i] = (bf16)p[i]; f[i + 8] = (bf16)p[16 + i]; }
  return f;
}

// B fragment (32x16 bf16): lane holds column n=l&15; K(e) = (e%8)+16*(e/8)+8*hi
// (same K order as the A fragment).
// rp points at the column's contiguous K-major data (row of W / row of Kmat / row of V^T) + k0.
DEV v16bf load_fragB_bf16(const bf16* rp, int hi) {
  const bf16* p = rp + 8 * hi;
  v8bf a0 = *(const v8bf*)(p);
  v8bf a1 = *(const v8bf*)(p + 16);
  v16bf f;
#pragma unroll
  for (int i = 0; i < 8; ++i) { f[i] = a0[i]; f[i + 8] = a1[i]; }
  return f;
}

// ---- kernel 0: f32 -> bf16 conversion -----------------------------------
__global__ __launch_bounds__(256) void cvt_bf16_kernel(const float* __restrict__ src,
                                                       bf16* __restrict__ dst, int n) {
  int i = blockIdx.x * 256 + threadIdx.x;
  if (i < n) dst[i] = (bf16)src[i];
}

// ---- kernel 1: QKV projection  X(3072x512) @ W^T(512x512) ---------------
// grid: (BS/16, NH, 3)   block: 32 (one wave). Each wave -> 16x64 strip.
__global__ __launch_bounds__(32) void qkv_proj_kernel(const bf16* __restrict__ encb,
                                                      const bf16* __restrict__ Wb,
                                                      bf16* __restrict__ Qb,
                                                      bf16* __restrict__ Kb,
                                                      bf16* __restrict__ Vtb) {
  const int rowTile = blockIdx.x;          // 0..191
  const int strip   = blockIdx.y;          // 0..7 == head (64 cols per head)
  const int which   = blockIdx.z;          // 0=Q 1=K 2=V
  const int lane = threadIdx.x, n = lane & 15, hi = lane >> 4;

  const bf16* W    = Wb + (size_t)which * HID_ * HID_;
  const bf16* arow = encb + (size_t)(rowTile * 16 + n) * HID_;

  v8f acc[4] = {};
  for (int k0 = 0; k0 < HID_; k0 += 32) {
    v16bf a = load_fragA_bf16(arow + k0, hi);
#pragma unroll
    for (int sub = 0; sub < 4; ++sub) {
      const bf16* brow = W + (size_t)(strip * 64 + sub * 16 + n) * HID_ + k0;
      acc[sub] = wmma_bf16(a, load_fragB_bf16(brow, hi), acc[sub]);
    }
  }
  const int h = strip;
#pragma unroll
  for (int sub = 0; sub < 4; ++sub) {
#pragma unroll
    for (int r = 0; r < 8; ++r) {
      int row = rowTile * 16 + r + 8 * hi;   // global (b*S+s) row
      int d   = sub * 16 + n;
      bf16 v  = (bf16)acc[sub][r];
      if (which == 0)      Qb [((size_t)h * BS_ + row) * HD_ + d] = v;
      else if (which == 1) Kb [((size_t)h * BS_ + row) * HD_ + d] = v;
      else                 Vtb[((size_t)h * HD_ + d) * BS_ + row] = v;  // V transposed
    }
  }
}

// ---- kernel 2: scores = Q K^T / sqrt(HID), causal mask, -> w region -----
// grid: (S/16, BS/64, B*NH)  block: 32.
__global__ __launch_bounds__(32) void scores_kernel(const bf16* __restrict__ Qb,
                                                    const bf16* __restrict__ Kb,
                                                    float* __restrict__ wout) {
  const int qTile  = blockIdx.x;           // 0..23
  const int kStrip = blockIdx.y;           // 0..47
  const int z = blockIdx.z;                // b*NH + h
  const int b = z >> 3, h = z & 7;
  const int lane = threadIdx.x, n = lane & 15, hi = lane >> 4;

  const bf16* arow = Qb + ((size_t)h * BS_ + b * S_ + qTile * 16 + n) * HD_;
  v8f acc[4] = {};
#pragma unroll
  for (int k0 = 0; k0 < HD_; k0 += 32) {
    v16bf a = load_fragA_bf16(arow + k0, hi);
#pragma unroll
    for (int sub = 0; sub < 4; ++sub) {
      const bf16* brow = Kb + ((size_t)h * BS_ + kStrip * 64 + sub * 16 + n) * HD_ + k0;
      acc[sub] = wmma_bf16(a, load_fragB_bf16(brow, hi), acc[sub]);
    }
  }
  float* wb = wout + (size_t)z * (B_ * SS_);
#pragma unroll
  for (int sub = 0; sub < 4; ++sub) {
    int j  = kStrip * 64 + sub * 16 + n;   // global key index (c*S + kk)
    int c  = j / S_, kk = j % S_;
#pragma unroll
    for (int r = 0; r < 8; ++r) {
      int q = qTile * 16 + r + 8 * hi;
      float v = (kk <= q) ? acc[sub][r] * INV_SCALE : -1.0e9f;
      wb[(size_t)c * SS_ + (size_t)q * S_ + kk] = v;
    }
  }
}

// ---- kernel 3: softmax over the combined (c,k) axis, in place -----------
// grid: B*NH*S rows, block: 256. 3072 elems/row -> 12 per thread in regs.
__global__ __launch_bounds__(256) void softmax_kernel(float* __restrict__ wout) {
  const int row = blockIdx.x;              // (b*NH+h)*S + q
  const int q  = row % S_;
  const int bh = row / S_;
  float* base = wout + (size_t)bh * (B_ * SS_) + (size_t)q * S_;
  const int tid = threadIdx.x;

  __shared__ float sh[256];
  float vals[12];
  float m = -1.0e30f;
#pragma unroll
  for (int t = 0; t < 12; ++t) {
    int i = tid + t * 256;
    float v = base[(size_t)(i / S_) * SS_ + (i % S_)];
    vals[t] = v;
    m = fmaxf(m, v);
  }
  sh[tid] = m; __syncthreads();
  for (int s = 128; s > 0; s >>= 1) {
    if (tid < s) sh[tid] = fmaxf(sh[tid], sh[tid + s]);
    __syncthreads();
  }
  const float M = sh[0]; __syncthreads();

  float ssum = 0.f;
#pragma unroll
  for (int t = 0; t < 12; ++t) { vals[t] = __expf(vals[t] - M); ssum += vals[t]; }
  sh[tid] = ssum; __syncthreads();
  for (int s = 128; s > 0; s >>= 1) {
    if (tid < s) sh[tid] += sh[tid + s];
    __syncthreads();
  }
  const float inv = 1.0f / sh[0];
#pragma unroll
  for (int t = 0; t < 12; ++t) {
    int i = tid + t * 256;
    base[(size_t)(i / S_) * SS_ + (i % S_)] = vals[t] * inv;
  }
}

// ---- kernel 4: ctx = w @ V  (per (b,h): (S x 3072) @ (3072 x 64)) -------
// grid: (S/16, B*NH)  block: 32.  w read as f32, converted to bf16 on load.
__global__ __launch_bounds__(32) void ctx_kernel(const float* __restrict__ wprob,
                                                 const bf16* __restrict__ Vtb,
                                                 bf16* __restrict__ ctxb) {
  const int qTile = blockIdx.x;            // 0..23
  const int z = blockIdx.y;                // b*NH + h
  const int b = z >> 3, h = z & 7;
  const int lane = threadIdx.x, n = lane & 15, hi = lane >> 4;

  const float* wrow = wprob + (size_t)z * (B_ * SS_) + (size_t)(qTile * 16 + n) * S_;
  const bf16*  vt   = Vtb + (size_t)h * HD_ * BS_;

  v8f acc[4] = {};
  for (int k0 = 0; k0 < BS_; k0 += 32) {
    int c = k0 / S_, kb = k0 % S_;         // 32 | 384, so one c per step
    v16bf a = load_fragA_f32(wrow + (size_t)c * SS_ + kb, hi);
#pragma unroll
    for (int sub = 0; sub < 4; ++sub) {
      const bf16* brow = vt + (size_t)(sub * 16 + n) * BS_ + k0;
      acc[sub] = wmma_bf16(a, load_fragB_bf16(brow, hi), acc[sub]);
    }
  }
#pragma unroll
  for (int sub = 0; sub < 4; ++sub) {
#pragma unroll
    for (int r = 0; r < 8; ++r) {
      int q = qTile * 16 + r + 8 * hi;
      int d = sub * 16 + n;
      ctxb[((size_t)(b * S_ + q)) * HID_ + h * HD_ + d] = (bf16)acc[sub][r];
    }
  }
}

// ---- kernel 5: out = ctx @ Wo^T + enc  (f32 result pre-LN) --------------
// grid: (BS/16, HID/64)  block: 32.
__global__ __launch_bounds__(32) void outproj_kernel(const bf16* __restrict__ ctxb,
                                                     const bf16* __restrict__ Wob,
                                                     const float* __restrict__ enc,
                                                     float* __restrict__ tmp) {
  const int rowTile = blockIdx.x;          // 0..191
  const int strip   = blockIdx.y;          // 0..7
  const int lane = threadIdx.x, n = lane & 15, hi = lane >> 4;

  const bf16* arow = ctxb + (size_t)(rowTile * 16 + n) * HID_;
  v8f acc[4] = {};
  for (int k0 = 0; k0 < HID_; k0 += 32) {
    v16bf a = load_fragA_bf16(arow + k0, hi);
#pragma unroll
    for (int sub = 0; sub < 4; ++sub) {
      const bf16* brow = Wob + (size_t)(strip * 64 + sub * 16 + n) * HID_ + k0;
      acc[sub] = wmma_bf16(a, load_fragB_bf16(brow, hi), acc[sub]);
    }
  }
#pragma unroll
  for (int sub = 0; sub < 4; ++sub) {
#pragma unroll
    for (int r = 0; r < 8; ++r) {
      int row = rowTile * 16 + r + 8 * hi;
      int col = strip * 64 + sub * 16 + n;
      tmp[(size_t)row * HID_ + col] = acc[sub][r] + enc[(size_t)row * HID_ + col];
    }
  }
}

// ---- kernel 6: LayerNorm over HID=512, -> final out ---------------------
__global__ __launch_bounds__(256) void layernorm_kernel(const float* __restrict__ tmp,
                                                        const float* __restrict__ gamma,
                                                        const float* __restrict__ beta,
                                                        float* __restrict__ out) {
  const int row = blockIdx.x;              // 0..3071
  const int tid = threadIdx.x;
  const float* x = tmp + (size_t)row * HID_;

  float v0 = x[tid], v1 = x[tid + 256];
  __shared__ float sh[256];
  sh[tid] = v0 + v1; __syncthreads();
  for (int s = 128; s > 0; s >>= 1) {
    if (tid < s) sh[tid] += sh[tid + s];
    __syncthreads();
  }
  const float mu = sh[0] * (1.0f / HID_); __syncthreads();
  float d0 = v0 - mu, d1 = v1 - mu;
  sh[tid] = d0 * d0 + d1 * d1; __syncthreads();
  for (int s = 128; s > 0; s >>= 1) {
    if (tid < s) sh[tid] += sh[tid + s];
    __syncthreads();
  }
  const float inv = rsqrtf(sh[0] * (1.0f / HID_) + EPS_);
  out[(size_t)row * HID_ + tid]       = d0 * inv * gamma[tid] + beta[tid];
  out[(size_t)row * HID_ + tid + 256] = d1 * inv * gamma[tid + 256] + beta[tid + 256];
}

// -------------------------------------------------------------------------
extern "C" void kernel_launch(void* const* d_in, const int* in_sizes, int n_in,
                              void* d_out, int out_size, void* d_ws, size_t ws_size,
                              hipStream_t stream) {
  (void)in_sizes; (void)n_in; (void)out_size; (void)ws_size;

  const float* enc   = (const float*)d_in[0];
  // d_in[1] = mask (tril), computed analytically instead
  const float* Wq    = (const float*)d_in[2];
  const float* Wk    = (const float*)d_in[3];
  const float* Wv    = (const float*)d_in[4];
  const float* Wo    = (const float*)d_in[5];
  const float* gamma = (const float*)d_in[6];
  const float* beta  = (const float*)d_in[7];

  float* out   = (float*)d_out;
  float* wprob = out + (size_t)BS_ * HID_;   // second output: (B,NH,B,S,S)

  // workspace carve-up (~24 MB)
  char* p = (char*)d_ws;
  bf16* encb = (bf16*)p; p += (size_t)BS_ * HID_ * sizeof(bf16);
  bf16* Wb   = (bf16*)p; p += (size_t)4 * HID_ * HID_ * sizeof(bf16); // Wq,Wk,Wv,Wo
  bf16* Qb   = (bf16*)p; p += (size_t)NH_ * BS_ * HD_ * sizeof(bf16);
  bf16* Kb   = (bf16*)p; p += (size_t)NH_ * BS_ * HD_ * sizeof(bf16);
  bf16* Vtb  = (bf16*)p; p += (size_t)NH_ * HD_ * BS_ * sizeof(bf16);
  bf16* ctxb = (bf16*)p; p += (size_t)BS_ * HID_ * sizeof(bf16);
  float* tmp = (float*)p;

  const int nEnc = BS_ * HID_;
  const int nW   = HID_ * HID_;
  cvt_bf16_kernel<<<(nEnc + 255) / 256, 256, 0, stream>>>(enc, encb, nEnc);
  cvt_bf16_kernel<<<(nW + 255) / 256, 256, 0, stream>>>(Wq, Wb + 0 * (size_t)nW, nW);
  cvt_bf16_kernel<<<(nW + 255) / 256, 256, 0, stream>>>(Wk, Wb + 1 * (size_t)nW, nW);
  cvt_bf16_kernel<<<(nW + 255) / 256, 256, 0, stream>>>(Wv, Wb + 2 * (size_t)nW, nW);
  cvt_bf16_kernel<<<(nW + 255) / 256, 256, 0, stream>>>(Wo, Wb + 3 * (size_t)nW, nW);

  qkv_proj_kernel<<<dim3(BS_ / 16, NH_, 3), 32, 0, stream>>>(encb, Wb, Qb, Kb, Vtb);
  scores_kernel<<<dim3(S_ / 16, BS_ / 64, B_ * NH_), 32, 0, stream>>>(Qb, Kb, wprob);
  softmax_kernel<<<B_ * NH_ * S_, 256, 0, stream>>>(wprob);
  ctx_kernel<<<dim3(S_ / 16, B_ * NH_), 32, 0, stream>>>(wprob, Vtb, ctxb);
  outproj_kernel<<<dim3(BS_ / 16, HID_ / 64), 32, 0, stream>>>(ctxb, Wb + 3 * (size_t)nW, enc, tmp);
  layernorm_kernel<<<BS_, 256, 0, stream>>>(tmp, gamma, beta, out);
}
